// GPT2QuantAttention_80358838108489
// MI455X (gfx1250) — 16-bit and 8-bit matrix operand order repaired and checked by reading, compiled, not run on hardware
//
#include <hip/hip_runtime.h>
#include <cstdint>

typedef _Float16 v16h __attribute__((ext_vector_type(16)));
typedef _Float16 v8h  __attribute__((ext_vector_type(8)));
typedef float    v8f  __attribute__((ext_vector_type(8)));
typedef int      v8i  __attribute__((ext_vector_type(8)));
typedef int      v4i  __attribute__((ext_vector_type(4)));
typedef int      v2i  __attribute__((ext_vector_type(2)));

constexpr int Bz = 2, S = 2048, D = 1024, H = 16, HD = 64, R = 8;
constexpr int M = Bz * S;
constexpr float LORA_SC = 16.0f / R;

static __device__ __forceinline__ v8f wmma_f16(v16h a, v16h b, v8f c) {
  return __builtin_amdgcn_wmma_f32_16x16x32_f16(false, a, false, b, (short)0, c, false, false);
}
static __device__ __forceinline__ v8i wmma_i8(v8i a, v8i b, v8i c) {
  return __builtin_amdgcn_wmma_i32_16x16x64_iu8(true, a, true, b, c, false, false);
}

__global__ void k_init(unsigned* p) { p[threadIdx.x] = 0u; }

__global__ void k_absmax(const float* __restrict__ x, int n, unsigned* slot) {
  __shared__ float red[256];
  int tid = threadIdx.x;
  float m = 0.0f;
  for (int i = blockIdx.x * blockDim.x + tid; i < n; i += gridDim.x * blockDim.x)
    m = fmaxf(m, fabsf(x[i]));
  red[tid] = m;
  __syncthreads();
  for (int s2 = 128; s2 > 0; s2 >>= 1) {
    if (tid < s2) red[tid] = fmaxf(red[tid], red[tid + s2]);
    __syncthreads();
  }
  if (tid == 0) atomicMax(slot, __float_as_uint(red[0]));
}

__global__ void k_quant(const float* __restrict__ x, int n, const unsigned* amax,
                        float* scale_out, int8_t* __restrict__ q) {
  float a = __uint_as_float(*amax);
  float s = fmaxf(a * (1.0f / 127.0f), 1e-8f);
  int i = blockIdx.x * blockDim.x + threadIdx.x;
  if (i == 0) *scale_out = s;
  if (i < n) {
    float v = rintf(x[i] / s);
    v = fminf(fmaxf(v, -127.0f), 127.0f);
    q[i] = (int8_t)v;
  }
}

__global__ void k_lora_xa(const float* __restrict__ x, const float* __restrict__ A,
                          float* __restrict__ t) {
  int idx = blockIdx.x * blockDim.x + threadIdx.x;
  int m = idx >> 3, r = idx & 7;
  const float* xr = x + (size_t)m * D;
  const float* ar = A + (size_t)r * D;
  float acc = 0.0f;
  for (int d = 0; d < D; d += 4) {
    float4 xv = *(const float4*)(xr + d);
    float4 av = *(const float4*)(ar + d);
    acc += xv.x * av.x + xv.y * av.y + xv.z * av.z + xv.w * av.w;
  }
  t[idx] = acc;
}

template <int MODE>
__global__ void __launch_bounds__(256) k_gemm_fused(const int8_t* __restrict__ A8,
                                                    const int8_t* __restrict__ W8,
                                                    const float* sA, const float* sW,
                                                    const float* __restrict__ bias,
                                                    const float* __restrict__ tLora,
                                                    const float* __restrict__ Bm,
                                                    float* __restrict__ outF,
                                                    _Float16* __restrict__ outH) {
  constexpr int K = 1024, N = 1024, NT = N / 64;
  int lane = threadIdx.x & 31;
  int gw   = (blockIdx.x * blockDim.x + threadIdx.x) >> 5;
  int tm = gw / NT, tn = gw % NT;
  int lo = lane & 15;
  int hi = lane >> 4;

  const int8_t* ap0 = A8 + (size_t)(tm * 32 + lo) * K + hi * 16;
  const int8_t* ap1 = ap0 + (size_t)16 * K;
  const int8_t* bp0 = W8 + (size_t)(tn * 64 + lo) * K + hi * 16;
  const int8_t* bp1 = bp0 + (size_t)16 * K;
  const int8_t* bp2 = bp0 + (size_t)32 * K;
  const int8_t* bp3 = bp0 + (size_t)48 * K;

  v8i acc[2][4] = {};
  for (int k0 = 0; k0 < K; k0 += 64) {
    v8i a0, a1, b0, b1, b2, b3;
    v2i* w;
    w = (v2i*)&a0;
    w[0] = *(const v2i*)(ap0 + k0);      w[1] = *(const v2i*)(ap0 + k0 + 8);
    w[2] = *(const v2i*)(ap0 + k0 + 32); w[3] = *(const v2i*)(ap0 + k0 + 40);
    w = (v2i*)&a1;
    w[0] = *(const v2i*)(ap1 + k0);      w[1] = *(const v2i*)(ap1 + k0 + 8);
    w[2] = *(const v2i*)(ap1 + k0 + 32); w[3] = *(const v2i*)(ap1 + k0 + 40);
    v4i* u;
    u = (v4i*)&b0; u[0] = *(const v4i*)(bp0 + k0); u[1] = *(const v4i*)(bp0 + k0 + 32);
    u = (v4i*)&b1; u[0] = *(const v4i*)(bp1 + k0); u[1] = *(const v4i*)(bp1 + k0 + 32);
    u = (v4i*)&b2; u[0] = *(const v4i*)(bp2 + k0); u[1] = *(const v4i*)(bp2 + k0 + 32);
    u = (v4i*)&b3; u[0] = *(const v4i*)(bp3 + k0); u[1] = *(const v4i*)(bp3 + k0 + 32);
    __builtin_prefetch(ap0 + k0 + 64, 0, 1);
    acc[0][0] = wmma_i8(a0, b0, acc[0][0]);
    acc[0][1] = wmma_i8(a0, b1, acc[0][1]);
    acc[0][2] = wmma_i8(a0, b2, acc[0][2]);
    acc[0][3] = wmma_i8(a0, b3, acc[0][3]);
    acc[1][0] = wmma_i8(a1, b0, acc[1][0]);
    acc[1][1] = wmma_i8(a1, b1, acc[1][1]);
    acc[1][2] = wmma_i8(a1, b2, acc[1][2]);
    acc[1][3] = wmma_i8(a1, b3, acc[1][3]);
  }

  float s = sA[0] * sW[0];
  int col[4];
  float bcol[4];
  float bl[4][8];
#pragma unroll
  for (int ni = 0; ni < 4; ni++) {
    col[ni] = tn * 64 + ni * 16 + lo;
    bcol[ni] = bias[col[ni]];
    const float* br = Bm + (size_t)col[ni] * R;
#pragma unroll
    for (int r = 0; r < R; r++) bl[ni][r] = br[r];
  }

#pragma unroll
  for (int mi = 0; mi < 2; mi++) {
#pragma unroll
    for (int i = 0; i < 8; i++) {
      int row = tm * 32 + mi * 16 + hi * 8 + i;
      const float* tr = tLora + (size_t)row * R;
      float t0 = tr[0], t1 = tr[1], t2 = tr[2], t3 = tr[3];
      float t4 = tr[4], t5 = tr[5], t6 = tr[6], t7 = tr[7];
#pragma unroll
      for (int ni = 0; ni < 4; ni++) {
        float l = t0 * bl[ni][0] + t1 * bl[ni][1] + t2 * bl[ni][2] + t3 * bl[ni][3] +
                  t4 * bl[ni][4] + t5 * bl[ni][5] + t6 * bl[ni][6] + t7 * bl[ni][7];
        float v = (float)acc[mi][ni][i] * s + bcol[ni] + LORA_SC * l;
        if (MODE == 0) {
          outF[(size_t)row * N + col[ni]] = v;
        } else {
          int b = row >> 11, sIdx = row & (S - 1);
          int h = col[ni] >> 6, hd = col[ni] & 63;
          size_t bh = (size_t)(b * H + h);
          if (MODE == 1) outH[(bh * S + sIdx) * HD + hd] = (_Float16)(v * 0.125f);
          if (MODE == 2) outH[(bh * S + sIdx) * HD + hd] = (_Float16)v;
          if (MODE == 3) outH[(bh * HD + hd) * S + sIdx] = (_Float16)v;
        }
      }
    }
  }
}

__global__ void __launch_bounds__(256) k_attention(const _Float16* __restrict__ qh,
                                                   const _Float16* __restrict__ kh,
                                                   const _Float16* __restrict__ vt,
                                                   const float* __restrict__ mask,
                                                   float* __restrict__ attn) {
  __shared__ _Float16 pst[8][16 * 32];
  int lane = threadIdx.x & 31;
  int wid  = threadIdx.x >> 5;
  int gw = blockIdx.x * 8 + wid;
  int qt = gw & (S / 16 - 1);
  int bh = gw / (S / 16);
  int b = bh >> 4, h = bh & 15;
  int q0 = qt * 16;
  int lo = lane & 15;
  int hi8 = (lane >> 4) * 8;
  int kb16 = (lane >> 4) * 16;

  const _Float16* qb  = qh + (size_t)bh * S * HD;
  const _Float16* kbp = kh + (size_t)bh * S * HD;
  const _Float16* vb  = vt + (size_t)bh * HD * S;

  v16h qf0, qf1;
  {
    const _Float16* qp = qb + (size_t)(q0 + lo) * HD + kb16;
    v8h* w0 = (v8h*)&qf0; v8h* w1 = (v8h*)&qf1;
    w0[0] = *(const v8h*)(qp);
    w0[1] = *(const v8h*)(qp + 8);
    w1[0] = *(const v8h*)(qp + 32);
    w1[1] = *(const v8h*)(qp + 40);
  }

  float m8[8], l8[8];
#pragma unroll
  for (int i = 0; i < 8; i++) { m8[i] = -3.0e38f; l8[i] = 0.0f; }
  v8f o0 = {}, o1 = {}, o2 = {}, o3 = {};
  const float* mrow = mask + (size_t)b * S;
  _Float16* ps = pst[wid];

  for (int j = 0; j < S; j += 32) {
    float mk0 = (1.0f - mrow[j + lo]) * -10000.0f;
    float mk1 = (1.0f - mrow[j + 16 + lo]) * -10000.0f;

    v16h k0g0 = *(const v16h*)(kbp + (size_t)(j + lo) * HD + kb16);
    v16h k0g1 = *(const v16h*)(kbp + (size_t)(j + lo) * HD + 32 + kb16);
    v16h k1g0 = *(const v16h*)(kbp + (size_t)(j + 16 + lo) * HD + kb16);
    v16h k1g1 = *(const v16h*)(kbp + (size_t)(j + 16 + lo) * HD + 32 + kb16);
    v16h vf0 = *(const v16h*)(vb + (size_t)(0  + lo) * S + j + kb16);
    v16h vf1 = *(const v16h*)(vb + (size_t)(16 + lo) * S + j + kb16);
    v16h vf2 = *(const v16h*)(vb + (size_t)(32 + lo) * S + j + kb16);
    v16h vf3 = *(const v16h*)(vb + (size_t)(48 + lo) * S + j + kb16);

    v8f s0 = {}, s1 = {};
    s0 = wmma_f16(qf0, k0g0, s0);
    s0 = wmma_f16(qf1, k0g1, s0);
    s1 = wmma_f16(qf0, k1g0, s1);
    s1 = wmma_f16(qf1, k1g1, s1);

#pragma unroll
    for (int i = 0; i < 8; i++) {
      float a0 = s0[i] + mk0;
      float a1 = s1[i] + mk1;
      float mx = fmaxf(a0, a1);
      mx = fmaxf(mx, __shfl_xor(mx, 1, 32));
      mx = fmaxf(mx, __shfl_xor(mx, 2, 32));
      mx = fmaxf(mx, __shfl_xor(mx, 4, 32));
      mx = fmaxf(mx, __shfl_xor(mx, 8, 32));
      float mn = fmaxf(m8[i], mx);
      float al = __expf(m8[i] - mn);
      float p0 = __expf(a0 - mn);
      float p1 = __expf(a1 - mn);
      float rs = p0 + p1;
      rs += __shfl_xor(rs, 1, 32);
      rs += __shfl_xor(rs, 2, 32);
      rs += __shfl_xor(rs, 4, 32);
      rs += __shfl_xor(rs, 8, 32);
      l8[i] = l8[i] * al + rs;
      m8[i] = mn;
      o0[i] *= al; o1[i] *= al; o2[i] *= al; o3[i] *= al;
      int row = i + hi8;
      ps[row * 32 + lo]      = (_Float16)p0;
      ps[row * 32 + 16 + lo] = (_Float16)p1;
    }
    asm volatile("s_wait_dscnt 0" ::: "memory");
    v16h pf;
    {
      v8h* pw = (v8h*)&pf;
      const _Float16* pp = ps + lo * 32 + kb16;
      pw[0] = *(const v8h*)(pp);
      pw[1] = *(const v8h*)(pp + 8);
    }
    o0 = wmma_f16(pf, vf0, o0);
    o1 = wmma_f16(pf, vf1, o1);
    o2 = wmma_f16(pf, vf2, o2);
    o3 = wmma_f16(pf, vf3, o3);
  }

#pragma unroll
  for (int i = 0; i < 8; i++) {
    float nv = 1.0f / l8[i];
    size_t m = (size_t)b * S + q0 + i + hi8;
    float* op = attn + m * D + (size_t)h * HD + lo;
    op[0]  = o0[i] * nv;
    op[16] = o1[i] * nv;
    op[32] = o2[i] * nv;
    op[48] = o3[i] * nv;
  }
}

extern "C" void kernel_launch(void* const* d_in, const int* in_sizes, int n_in,
                              void* d_out, int out_size, void* d_ws, size_t ws_size,
                              hipStream_t stream) {
  (void)in_sizes; (void)n_in; (void)out_size; (void)ws_size;
  const float* hidden = (const float*)d_in[0];
  const float* mask   = (const float*)d_in[1];
  const float* Wm[4] = {(const float*)d_in[2], (const float*)d_in[6],
                        (const float*)d_in[10], (const float*)d_in[14]};
  const float* bv[4] = {(const float*)d_in[3], (const float*)d_in[7],
                        (const float*)d_in[11], (const float*)d_in[15]};
  const float* Am[4] = {(const float*)d_in[4], (const float*)d_in[8],
                        (const float*)d_in[12], (const float*)d_in[16]};
  const float* Bm[4] = {(const float*)d_in[5], (const float*)d_in[9],
                        (const float*)d_in[13], (const float*)d_in[17]};

  char* w = (char*)d_ws;
  unsigned* amax = (unsigned*)w;
  float* scl = (float*)(w + 256);
  size_t off = 1024;
  int8_t* x8 = (int8_t*)(w + off); off += (size_t)M * D;
  int8_t* w8 = (int8_t*)(w + off); off += 4 * (size_t)D * D;
  float* tB  = (float*)(w + off);  off += 4 * (size_t)M * R * 4;
  float* at  = (float*)(w + off);  off += (size_t)M * D * 4;
  _Float16* qhh = (_Float16*)(w + off); off += (size_t)M * D * 2;
  _Float16* khh = (_Float16*)(w + off); off += (size_t)M * D * 2;
  _Float16* vth = (_Float16*)(w + off); off += (size_t)M * D * 2;

  constexpr int GEMM_BLOCKS = (M / 32) * (D / 64) / 8;

  k_init<<<1, 256, 0, stream>>>((unsigned*)w);
  k_absmax<<<1024, 256, 0, stream>>>(hidden, M * D, amax + 0);
  for (int p = 0; p < 4; p++)
    k_absmax<<<512, 256, 0, stream>>>(Wm[p], D * D, amax + 1 + p);
  k_quant<<<(M * D) / 256, 256, 0, stream>>>(hidden, M * D, amax + 0, scl + 0, x8);
  for (int p = 0; p < 4; p++)
    k_quant<<<(D * D) / 256, 256, 0, stream>>>(Wm[p], D * D, amax + 1 + p, scl + 1 + p,
                                               w8 + (size_t)p * D * D);
  for (int p = 0; p < 3; p++)
    k_lora_xa<<<(M * R) / 256, 256, 0, stream>>>(hidden, Am[p], tB + (size_t)p * M * R);

  k_gemm_fused<1><<<GEMM_BLOCKS, 256, 0, stream>>>(x8, w8 + 0 * (size_t)D * D, scl + 0,
                                                   scl + 1, bv[0], tB + 0 * (size_t)M * R,
                                                   Bm[0], nullptr, qhh);
  k_gemm_fused<2><<<GEMM_BLOCKS, 256, 0, stream>>>(x8, w8 + 1 * (size_t)D * D, scl + 0,
                                                   scl + 2, bv[1], tB + 1 * (size_t)M * R,
                                                   Bm[1], nullptr, khh);
  k_gemm_fused<3><<<GEMM_BLOCKS, 256, 0, stream>>>(x8, w8 + 2 * (size_t)D * D, scl + 0,
                                                   scl + 3, bv[2], tB + 2 * (size_t)M * R,
                                                   Bm[2], nullptr, vth);

  k_attention<<<(Bz * H * (S / 16)) / 8, 256, 0, stream>>>(qhh, khh, vth, mask, at);

  k_absmax<<<1024, 256, 0, stream>>>(at, M * D, amax + 5);
  k_quant<<<(M * D) / 256, 256, 0, stream>>>(at, M * D, amax + 5, scl + 5, x8);
  k_lora_xa<<<(M * R) / 256, 256, 0, stream>>>(at, Am[3], tB + 3 * (size_t)M * R);
  k_gemm_fused<0><<<GEMM_BLOCKS, 256, 0, stream>>>(x8, w8 + 3 * (size_t)D * D, scl + 5,
                                                   scl + 4, bv[3], tB + 3 * (size_t)M * R,
                                                   Bm[3], (float*)d_out, nullptr);
}
